// Net_2327872274909
// MI455X (gfx1250) — matrix operand order repaired and checked by reading, not compiled, not run on hardware
//
#include <hip/hip_runtime.h>
#include <stdint.h>

// ---------------------------------------------------------------------------
// EdgeConv GNN forward for MI455X (gfx1250, wave32, WMMA bf16 -> f32 accum)
// ---------------------------------------------------------------------------

typedef __attribute__((ext_vector_type(16))) __bf16 bf16x16;
typedef __attribute__((ext_vector_type(8)))  float  f32x8;

#define WPB 4           // waves per block in the edge kernel
#define DIN 64          // all layers have input dim 64 (F_IN=64, DIMS[0..2]=64)
#define NNODES_MAX 0    // unused

__device__ __forceinline__ float bf2f(unsigned short u) {
    union { unsigned int i; float f; } c; c.i = ((unsigned int)u) << 16; return c.f;
}
__device__ __forceinline__ unsigned short f2bf(float f) {
    union { float f; unsigned int i; } c; c.f = f;
    unsigned int i = c.i;
    unsigned int r = i + 0x7FFFu + ((i >> 16) & 1u);   // round to nearest even
    return (unsigned short)(r >> 16);
}
__device__ __forceinline__ unsigned int pack2bf(float a, float b) {
    return (unsigned int)f2bf(a) | ((unsigned int)f2bf(b) << 16);
}
__device__ __forceinline__ float mishf(float x) {
    float sp = (x > 20.0f) ? x : log1pf(__expf(x));
    return x * tanhf(sp);
}
// order-independent float max via sign-aware integer atomics (deterministic)
__device__ __forceinline__ void atomicMaxF(float* addr, float v) {
    if (v >= 0.0f) atomicMax((int*)addr, __float_as_int(v));
    else           atomicMin((unsigned int*)addr, __float_as_uint(v));
}

// ---------------------------------------------------------------------------
// Small utility kernels
// ---------------------------------------------------------------------------
__global__ void cvt_f32_bf16(const float* __restrict__ x,
                             unsigned short* __restrict__ xbf, int total) {
    int i = blockIdx.x * blockDim.x + threadIdx.x;
    if (i < total) xbf[i] = f2bf(x[i]);
}

__global__ void init_neg_inf(unsigned int* __restrict__ p, int total) {
    int i = blockIdx.x * blockDim.x + threadIdx.x;
    if (i < total) p[i] = 0xFF800000u;   // -inf
}

// pack float weight [Kreal, Nreal] into WMMA B-fragment tiles:
// tile t = kc*NT + nt ; per tile 32 lanes x 16 bf16 ;
// lane element i -> k = kc*32 + (lane<16?0:16) + i , n = nt*16 + (lane&15)
__global__ void pack_w(const float* __restrict__ w, unsigned short* __restrict__ p,
                       int Kreal, int Nreal, int NT, int total) {
    int idx = blockIdx.x * blockDim.x + threadIdx.x;
    if (idx >= total) return;
    int i    = idx & 15;
    int lane = (idx >> 4) & 31;
    int t    = idx >> 9;
    int nt   = t % NT;
    int kc   = t / NT;
    int k = kc * 32 + ((lane < 16) ? 0 : 16) + i;
    int n = nt * 16 + (lane & 15);
    float v = (k < Kreal && n < Nreal) ? w[(size_t)k * Nreal + n] : 0.0f;
    p[idx] = f2bf(v);
}

__global__ void fix_inplace(float* __restrict__ x, int total) {
    int i = blockIdx.x * blockDim.x + threadIdx.x;
    if (i >= total) return;
    unsigned int b = __float_as_uint(x[i]);
    if ((b & 0x7F800000u) == 0x7F800000u) x[i] = 0.0f;   // inf/nan -> 0
}

__global__ void fix_to_out(float* __restrict__ out, const float* __restrict__ agg,
                           int total) {
    int i = blockIdx.x * blockDim.x + threadIdx.x;
    if (i >= total) return;
    float v = agg[i];
    unsigned int b = __float_as_uint(v);
    out[i] = ((b & 0x7F800000u) == 0x7F800000u) ? 0.0f : v;
}

// deterministic per-feature mean/var: one block per feature column
__global__ void bn_stats(const float* __restrict__ x, float* __restrict__ stats,
                         int N, int dout) {
    __shared__ float ss[256], sq[256];
    int f = blockIdx.x;
    float s = 0.0f, q = 0.0f;
    for (int i = threadIdx.x; i < N; i += 256) {
        float v = x[(size_t)i * dout + f];
        s += v; q += v * v;
    }
    ss[threadIdx.x] = s; sq[threadIdx.x] = q;
    __syncthreads();
    for (int o = 128; o > 0; o >>= 1) {
        if (threadIdx.x < o) {
            ss[threadIdx.x] += ss[threadIdx.x + o];
            sq[threadIdx.x] += sq[threadIdx.x + o];
        }
        __syncthreads();
    }
    if (threadIdx.x == 0) {
        float mu = ss[0] / (float)N;
        stats[f]        = mu;
        stats[dout + f] = sq[0] / (float)N - mu * mu;
    }
}

// apply BN and emit next-layer bf16 activations
__global__ void bn_apply_cvt(const float* __restrict__ x, const float* __restrict__ stats,
                             const float* __restrict__ g, const float* __restrict__ be,
                             unsigned short* __restrict__ xbf, int total, int dout) {
    int i = blockIdx.x * blockDim.x + threadIdx.x;
    if (i >= total) return;
    int f = i % dout;
    float mu = stats[f], var = stats[dout + f];
    float v = (x[i] - mu) * rsqrtf(var + 1e-5f) * g[f] + be[f];
    xbf[i] = f2bf(v);
}

// ---------------------------------------------------------------------------
// Fused per-edge MLP + segment-max kernel.
//   One wave32 -> 16 edges. GEMM1: [16x128]x[128,DOUT]; mish;
//   GEMM2: [16x(NT1*16)]x[.,DOUT]; atomic segment-max into agg.
// Templated so all WMMA loops fully unroll (accumulators in fixed VGPRs).
// ---------------------------------------------------------------------------
template<int DOUT, int NT1, int KC2, int NT2>
__global__ __launch_bounds__(WPB * 32)
void edge_kernel(const unsigned short* __restrict__ xbf,     // [N, 64] bf16
                 const int* __restrict__ src, const int* __restrict__ dst,
                 const unsigned short* __restrict__ w1p,     // packed B tiles
                 const unsigned short* __restrict__ w2p,
                 const float* __restrict__ b1, const float* __restrict__ b2,
                 float* __restrict__ agg,                    // [N, DOUT] f32
                 int E) {
    __shared__ unsigned int mbuf[WPB][16 * 64];   // 16 rows x 128 bf16 (dwords)
    __shared__ unsigned int hbuf[WPB][16 * 32];   // 16 rows x  64 bf16 (dwords)
    __shared__ int          dbuf[WPB][16];

    const int lane = threadIdx.x & 31;
    const int wv   = threadIdx.x >> 5;
    const int tile = blockIdx.x * WPB + wv;
    const int eb   = tile * 16;
    if (eb >= E) return;                 // uniform per wave -> EXEC stays full

    unsigned int* mrow = mbuf[wv];
    unsigned int* hrow = hbuf[wv];

    // ---- Stage A: build message tile m = [xi | xj - xi] in LDS (bf16) ----
    {
        int er   = lane >> 1;            // edge row 0..15
        int half = lane & 1;             // feature half: 0 -> [0,32), 1 -> [32,64)
        int e = eb + er; if (e >= E) e = E - 1;   // dup last edge: max-idempotent
        int s = src[e];
        int d = dst[e];
        if (half == 0) dbuf[wv][er] = d;
        const uint4* xi4 = (const uint4*)(xbf + (size_t)d * DIN + half * 32);
        const uint4* xj4 = (const uint4*)(xbf + (size_t)s * DIN + half * 32);
        #pragma unroll
        for (int c = 0; c < 4; ++c) {
            uint4 a = xi4[c];
            uint4 b = xj4[c];
            unsigned int ia[4] = { a.x, a.y, a.z, a.w };
            unsigned int jb[4] = { b.x, b.y, b.z, b.w };
            unsigned int df[4];
            #pragma unroll
            for (int q = 0; q < 4; ++q) {
                float i0 = bf2f((unsigned short)(ia[q] & 0xFFFFu));
                float i1 = bf2f((unsigned short)(ia[q] >> 16));
                float j0 = bf2f((unsigned short)(jb[q] & 0xFFFFu));
                float j1 = bf2f((unsigned short)(jb[q] >> 16));
                df[q] = pack2bf(j0 - i0, j1 - i1);
            }
            int colXi = half * 32 + c * 8;        // bf16 index within 128-wide row
            int colDf = 64 + colXi;
            *(uint4*)&mrow[er * 64 + (colXi >> 1)] = a;
            *(uint4*)&mrow[er * 64 + (colDf >> 1)] =
                make_uint4(df[0], df[1], df[2], df[3]);
        }
    }
    // zero-pad h K-range when DOUT < 16 (layer 3: cols 16..31 must be zero)
    if (NT1 * 16 < KC2 * 32) {
        for (int t = lane; t < 16 * 8; t += 32) {
            int r = t >> 3, cdw = 8 + (t & 7);
            hrow[r * 32 + cdw] = 0u;
        }
    }
    __builtin_amdgcn_wave_barrier();

    const int arow = lane & 15;                  // A-fragment row
    const int klo8 = (lane < 16) ? 0 : 16;       // A-fragment K offset (bf16 units)
    const int mhalf = (lane < 16) ? 0 : 8;       // C/D row offset

    // ---- GEMM1: m[16x128] x w1[128 x NT1*16] ----
    f32x8 c1[NT1];
    #pragma unroll
    for (int nt = 0; nt < NT1; ++nt) c1[nt] = (f32x8)(0.0f);

    #pragma unroll
    for (int kc = 0; kc < 4; ++kc) {
        union { uint4 u[2]; bf16x16 v; } A;
        int base = arow * 64 + ((kc * 32 + klo8) >> 1);
        A.u[0] = *(const uint4*)&mrow[base];
        A.u[1] = *(const uint4*)&mrow[base + 4];    // +8 bf16
        #pragma unroll
        for (int nt = 0; nt < NT1; ++nt) {
            union { uint4 u[2]; bf16x16 v; } B;
            const uint4* bp =
                (const uint4*)(w1p + ((size_t)(kc * NT1 + nt) * 32 + lane) * 16);
            B.u[0] = bp[0]; B.u[1] = bp[1];
            c1[nt] = __builtin_amdgcn_wmma_f32_16x16x32_bf16(
                false, A.v, false, B.v, (short)0, c1[nt], false, false);
        }
    }

    // ---- bias + mish, stage h into LDS as bf16 ----
    #pragma unroll
    for (int nt = 0; nt < NT1; ++nt) {
        int n = nt * 16 + (lane & 15);
        float bias = (n < DOUT) ? b1[n] : 0.0f;
        #pragma unroll
        for (int r = 0; r < 8; ++r) {
            int row = r + mhalf;
            float h = mishf(c1[nt][r] + bias);
            ((unsigned short*)hrow)[row * 64 + n] = f2bf(h);
        }
    }
    __builtin_amdgcn_wave_barrier();

    // ---- GEMM2: h[16 x KC2*32] x w2[KC2*32 x NT2*16] ----
    f32x8 c2[NT2];
    #pragma unroll
    for (int nt = 0; nt < NT2; ++nt) c2[nt] = (f32x8)(0.0f);

    #pragma unroll
    for (int kc = 0; kc < KC2; ++kc) {
        union { uint4 u[2]; bf16x16 v; } A;
        int base = arow * 32 + ((kc * 32 + klo8) >> 1);
        A.u[0] = *(const uint4*)&hrow[base];
        A.u[1] = *(const uint4*)&hrow[base + 4];
        #pragma unroll
        for (int nt = 0; nt < NT2; ++nt) {
            union { uint4 u[2]; bf16x16 v; } B;
            const uint4* bp =
                (const uint4*)(w2p + ((size_t)(kc * NT2 + nt) * 32 + lane) * 16);
            B.u[0] = bp[0]; B.u[1] = bp[1];
            c2[nt] = __builtin_amdgcn_wmma_f32_16x16x32_bf16(
                false, A.v, false, B.v, (short)0, c2[nt], false, false);
        }
    }

    // ---- bias + segment max (atomic, order-independent) ----
    #pragma unroll
    for (int nt = 0; nt < NT2; ++nt) {
        int n = nt * 16 + (lane & 15);
        if (n < DOUT) {
            float bias = b2[n];
            #pragma unroll
            for (int r = 0; r < 8; ++r) {
                int row = r + mhalf;
                int dn = dbuf[wv][row];
                atomicMaxF(&agg[(size_t)dn * DOUT + n], c2[nt][r] + bias);
            }
        }
    }
}

// ---------------------------------------------------------------------------
// Host orchestration
// ---------------------------------------------------------------------------
static inline int ceildiv(int a, int b) { return (a + b - 1) / b; }

extern "C" void kernel_launch(void* const* d_in, const int* in_sizes, int n_in,
                              void* d_out, int out_size, void* d_ws, size_t ws_size,
                              hipStream_t stream) {
    (void)in_sizes; (void)n_in; (void)ws_size;
    const int N = 50000, E = 800000;

    const float* x0  = (const float*)d_in[0];
    const int*   ei  = (const int*)d_in[1];      // [2, E]
    const int*   src = ei;
    const int*   dst = ei + E;
    // per-layer params: base 3, layers 0..2 have 6 tensors, layer 3 has 4
    const float* w1[4]; const float* b1[4]; const float* w2[4]; const float* b2[4];
    const float* g[3];  const float* be[3];
    for (int l = 0; l < 4; ++l) {
        int base = 3 + 6 * l;
        w1[l] = (const float*)d_in[base + 0];
        b1[l] = (const float*)d_in[base + 1];
        w2[l] = (const float*)d_in[base + 2];
        b2[l] = (const float*)d_in[base + 3];
        if (l < 3) { g[l] = (const float*)d_in[base + 4]; be[l] = (const float*)d_in[base + 5]; }
    }
    const int douts[4] = { 64, 64, 64, 8 };
    const int NT1s[4]  = { 4, 4, 4, 1 };
    const int KC2s[4]  = { 2, 2, 2, 1 };

    // workspace layout
    uintptr_t p = (uintptr_t)d_ws;
    float*          agg  = (float*)p;              p += (size_t)N * 64 * 4;   // 12.8 MB
    unsigned short* xbf  = (unsigned short*)p;     p += (size_t)N * 64 * 2;   //  6.4 MB
    unsigned short* w1p[4]; for (int l = 0; l < 4; ++l) { w1p[l] = (unsigned short*)p; p += 16384; }
    unsigned short* w2p[4]; for (int l = 0; l < 4; ++l) { w2p[l] = (unsigned short*)p; p += 8192;  }
    float*          stats = (float*)p;             p += 512;

    // initial bf16 activations
    cvt_f32_bf16<<<ceildiv(N * 64, 256), 256, 0, stream>>>(x0, xbf, N * 64);

    // pack weights into WMMA B-fragment order (zero padded)
    for (int l = 0; l < 4; ++l) {
        int t1 = 4 * NT1s[l] * 512;                     // KC1=4 always (K=128)
        pack_w<<<ceildiv(t1, 256), 256, 0, stream>>>(w1[l], w1p[l], 128, douts[l], NT1s[l], t1);
        int K2 = (l < 3) ? 64 : 8;
        int t2 = KC2s[l] * NT1s[l] * 512;               // NT2 == NT1
        pack_w<<<ceildiv(t2, 256), 256, 0, stream>>>(w2[l], w2p[l], K2, douts[l], NT1s[l], t2);
    }

    const int tiles  = ceildiv(E, 16);
    const int eblks  = ceildiv(tiles, WPB);

    for (int l = 0; l < 4; ++l) {
        const int dout = douts[l];
        const int tot  = N * dout;
        init_neg_inf<<<ceildiv(tot, 256), 256, 0, stream>>>((unsigned int*)agg, tot);

        if (l < 3) {
            edge_kernel<64, 4, 2, 4><<<eblks, WPB * 32, 0, stream>>>(
                xbf, src, dst, w1p[l], w2p[l], b1[l], b2[l], agg, E);
            fix_inplace<<<ceildiv(tot, 256), 256, 0, stream>>>(agg, tot);
            bn_stats<<<dout, 256, 0, stream>>>(agg, stats, N, dout);
            bn_apply_cvt<<<ceildiv(tot, 256), 256, 0, stream>>>(
                agg, stats, g[l], be[l], xbf, tot, dout);
        } else {
            edge_kernel<8, 1, 1, 1><<<eblks, WPB * 32, 0, stream>>>(
                xbf, src, dst, w1p[l], w2p[l], b1[l], b2[l], agg, E);
            fix_to_out<<<ceildiv(tot, 256), 256, 0, stream>>>((float*)d_out, agg, tot);
        }
    }
}
